// MoELayer_49091476193630
// MI455X (gfx1250) — compile-verified, matrix operand order checked by reading, not run on hardware
//
#include <hip/hip_runtime.h>
#include <hip/hip_bf16.h>
#include <cstdint>

// ---------------------------------------------------------------------------
// MoE (top-2 of 8 experts, SwiGLU) for MI455X / gfx1250.
// router -> gather tokens per expert -> bf16 WMMA GEMMs with double-buffered
// LDS tiles fed by GLOBAL_LOAD_ASYNC_TO_LDS (ASYNCcnt) -> weighted combine.
// ---------------------------------------------------------------------------

typedef unsigned short u16;
typedef __attribute__((ext_vector_type(16))) __bf16 v16bf;
typedef __attribute__((ext_vector_type(8)))  float  v8f;
typedef __attribute__((ext_vector_type(8)))  u16    u16x8;
typedef __attribute__((ext_vector_type(16))) u16    u16x16;
typedef __attribute__((ext_vector_type(4)))  int    i32x4;

static constexpr int N_TOK = 4096;   // B*T
static constexpr int DDIM  = 1024;   // model dim
static constexpr int HDIM  = 4096;   // hidden dim
static constexpr int NEXP  = 8;      // experts
static constexpr int TOPK  = 2;

static constexpr int BM = 128;       // token rows per block
static constexpr int BN = 64;        // output cols per block
static constexpr int BK = 32;        // K chunk (one bf16 WMMA K)
static constexpr int LDSTR = BK + 8; // padded LDS row stride (u16), 80B rows (16B aligned)

// ---- CDNA5 async global->LDS path (guarded; falls back to ds_store) -------
#if defined(__gfx1250__) && \
    __has_builtin(__builtin_amdgcn_global_load_async_to_lds_b128) && \
    __has_builtin(__builtin_amdgcn_s_wait_asynccnt)
#define HAVE_ASYNC_LDS 1
#else
#define HAVE_ASYNC_LDS 0
#endif

#if HAVE_ASYNC_LDS
typedef __attribute__((address_space(1))) i32x4 gi32x4;   // global int4
typedef __attribute__((address_space(3))) i32x4 li32x4;   // LDS int4
#endif

__device__ __forceinline__ void cp_b128(u16* lds, const u16* glob) {
#if HAVE_ASYNC_LDS
  __builtin_amdgcn_global_load_async_to_lds_b128(
      (gi32x4*)(u16*)glob, (li32x4*)lds, /*offset=*/0, /*cpol=*/0);
#else
  *(u16x8*)lds = *(const u16x8*)glob;
#endif
}

__device__ __forceinline__ void async_wait_all() {
#if HAVE_ASYNC_LDS
  __builtin_amdgcn_s_wait_asynccnt(0);
#endif
}

__device__ __forceinline__ u16 f32_to_bf16(float f) {
  unsigned u = __builtin_bit_cast(unsigned, f);
  unsigned r = (u + 0x7FFFu + ((u >> 16) & 1u)) >> 16;   // round-to-nearest-even
  return (u16)r;
}

// Per-lane WMMA fragment load, CDNA5 16-bit A/B layout:
//   lanes 0-15 : row/col = lane,    elems 0-7 = K 0..7,  elems 8-15 = K 16..23
//   lanes16-31 : row/col = lane-16, elems 0-7 = K 8..15, elems 8-15 = K 24..31
__device__ __forceinline__ v16bf load_frag(const u16* rowBase, int lane) {
  int kk = (lane >> 4) << 3;                    // 0 or 8
  u16x8 lo = *(const u16x8*)(rowBase + kk);
  u16x8 hi = *(const u16x8*)(rowBase + kk + 16);
  u16x16 u = __builtin_shufflevector(lo, hi, 0,1,2,3,4,5,6,7,8,9,10,11,12,13,14,15);
  return __builtin_bit_cast(v16bf, u);
}

__device__ __forceinline__ v8f wmma_bf16(v16bf a, v16bf b, v8f c) {
  return __builtin_amdgcn_wmma_f32_16x16x32_bf16(
      false, a, false, b, (short)0, c, false, false);
}

// ---------------------------------------------------------------------------
// Small helper kernels
// ---------------------------------------------------------------------------
__global__ void zero_counts_kernel(int* counts) {
  if (threadIdx.x < NEXP) counts[threadIdx.x] = 0;
}

__global__ void f2bf_kernel(const float* __restrict__ src, u16* __restrict__ dst, int n) {
  int i = blockIdx.x * blockDim.x + threadIdx.x;
  int stride = gridDim.x * blockDim.x;
  for (; i < n; i += stride) dst[i] = f32_to_bf16(src[i]);
}

// One wave32 per token: 8 length-1024 dot products, cross-lane reduce, top-2
// (jax tie-break: lowest index wins), softmax over the two selected logits.
__global__ __launch_bounds__(256) void router_kernel(
    const float* __restrict__ x, const float* __restrict__ gate_w,
    const float* __restrict__ rscale,
    int* counts, int* tok_e, int* tok_pos, float* tok_w) {
  int lane = threadIdx.x & 31;
  int n = blockIdx.x * 8 + (threadIdx.x >> 5);
  if (n >= N_TOK) return;
  float acc[NEXP] = {};
  const float* xr = x + (size_t)n * DDIM;
  for (int d = lane; d < DDIM; d += 32) {
    float xv = xr[d];
#pragma unroll
    for (int e = 0; e < NEXP; ++e) acc[e] += xv * gate_w[e * DDIM + d];
  }
#pragma unroll
  for (int e = 0; e < NEXP; ++e)
#pragma unroll
    for (int off = 16; off; off >>= 1) acc[e] += __shfl_xor(acc[e], off, 32);
  if (lane == 0) {
    float s = rscale[0];
    float lg[NEXP];
#pragma unroll
    for (int e = 0; e < NEXP; ++e) lg[e] = acc[e] * s;
    int e0 = 0;
#pragma unroll
    for (int e = 1; e < NEXP; ++e) if (lg[e] > lg[e0]) e0 = e;
    int e1 = (e0 == 0) ? 1 : 0;
#pragma unroll
    for (int e = 0; e < NEXP; ++e) if (e != e0 && lg[e] > lg[e1]) e1 = e;
    float t  = __expf(lg[e1] - lg[e0]);     // <= 1
    float w0 = __builtin_amdgcn_rcpf(1.0f + t);
    float w1 = t * w0;
    int p0 = atomicAdd(&counts[e0], 1);
    int p1 = atomicAdd(&counts[e1], 1);
    tok_e[n * 2]     = e0; tok_pos[n * 2]     = p0; tok_w[n * 2]     = w0;
    tok_e[n * 2 + 1] = e1; tok_pos[n * 2 + 1] = p1; tok_w[n * 2 + 1] = w1;
  }
}

__global__ void offsets_kernel(const int* counts, int* offsets) {
  if (threadIdx.x == 0 && blockIdx.x == 0) {
    int s = 0;
    for (int e = 0; e < NEXP; ++e) { offsets[e] = s; s += counts[e]; }
  }
}

__global__ void assign_slots_kernel(const int* offsets, const int* tok_e,
                                    const int* tok_pos, int* tok_list, int* slot_of) {
  int idx = blockIdx.x * blockDim.x + threadIdx.x;
  if (idx >= N_TOK * TOPK) return;
  int slot = offsets[tok_e[idx]] + tok_pos[idx];
  tok_list[slot] = idx >> 1;     // token id
  slot_of[idx]   = slot;
}

// ---------------------------------------------------------------------------
// Stage 1: per expert, gathered tokens:  h = silu(X@W1^T + b1) * (X@W2^T + b2)
// 128 tokens x 64 hidden cols per block; 8 waves each own a 32x32 sub-tile of
// BOTH products (8 WMMAs / wave / K-chunk). Double-buffered async LDS tiles.
// ---------------------------------------------------------------------------
__global__ __launch_bounds__(256) void moe_stage1(
    const u16* __restrict__ xb, const u16* __restrict__ w1b,
    const u16* __restrict__ w2b, const float* __restrict__ b1,
    const float* __restrict__ b2, const int* __restrict__ counts,
    const int* __restrict__ offsets, const int* __restrict__ tok_list,
    u16* __restrict__ h_act) {
  const int e  = blockIdx.z;
  const int ne = counts[e];
  const int ti = blockIdx.y;
  if (ti * BM >= ne) return;
  const int hTile = blockIdx.x * BN;
  const int seg   = offsets[e];

  __shared__ u16 As [2][BM * LDSTR];
  __shared__ u16 B1s[2][BN * LDSTR];
  __shared__ u16 B2s[2][BN * LDSTR];

  const int tid  = threadIdx.x;
  const int lane = tid & 31;
  const int wm   = (tid >> 5) & 3;   // wave row group (4 x 32 rows)
  const int wn   = (tid >> 5) >> 2;  // wave col group (2 x 32 cols)

  // A loader: 128 rows x 32 K, 16 u16 per thread (2 x b128)
  const int arow = tid >> 1;
  const int aseg = (tid & 1) << 4;
  const int gr   = ti * BM + arow;
  const int tok  = tok_list[seg + (gr < ne ? gr : ne - 1)];
  const u16* aG  = xb + (size_t)tok * DDIM + aseg;
  const int aoff = arow * LDSTR + aseg;

  // B loaders: threads 0-127 -> W1, 128-255 -> W2; 64 rows x 32 K
  const int brow = (tid & 127) >> 1;
  const int bseg = (tid & 1) << 4;
  const bool isB1 = tid < 128;
  const u16* bG  = (isB1 ? w1b : w2b) +
                   ((size_t)e * HDIM + (hTile + brow)) * DDIM + bseg;
  const int boff = brow * LDSTR + bseg;

  v8f acc1[2][2] = {};
  v8f acc2[2][2] = {};

  auto load_tiles = [&](int buf, int k0) {
    const u16* a = aG + k0;
    u16* al = As[buf] + aoff;
    cp_b128(al, a); cp_b128(al + 8, a + 8);
    const u16* b = bG + k0;
    u16* bl = (isB1 ? B1s[buf] : B2s[buf]) + boff;
    cp_b128(bl, b); cp_b128(bl + 8, b + 8);
  };

  auto compute_chunk = [&](int buf) {
    const u16* Ab  = As [buf] + (wm * 32 + (lane & 15)) * LDSTR;
    const u16* B1b = B1s[buf] + (wn * 32 + (lane & 15)) * LDSTR;
    const u16* B2b = B2s[buf] + (wn * 32 + (lane & 15)) * LDSTR;
    v16bf bf1[2], bf2[2];
#pragma unroll
    for (int ni = 0; ni < 2; ++ni) {
      bf1[ni] = load_frag(B1b + ni * 16 * LDSTR, lane);
      bf2[ni] = load_frag(B2b + ni * 16 * LDSTR, lane);
    }
#pragma unroll
    for (int mi = 0; mi < 2; ++mi) {
      v16bf af = load_frag(Ab + mi * 16 * LDSTR, lane);
#pragma unroll
      for (int ni = 0; ni < 2; ++ni) {
        acc1[mi][ni] = wmma_bf16(af, bf1[ni], acc1[mi][ni]);
        acc2[mi][ni] = wmma_bf16(af, bf2[ni], acc2[mi][ni]);
      }
    }
  };

  constexpr int NK = DDIM / BK;   // 32 chunks
  load_tiles(0, 0);
  for (int kc = 0; kc < NK; ++kc) {
    int cur = kc & 1;
    async_wait_all();
    __syncthreads();
    if (kc + 1 < NK) load_tiles(cur ^ 1, (kc + 1) * BK);
    compute_chunk(cur);
  }

  // Epilogue: C layout -> row = r + 8*(lane>=16), col = lane&15
  const int rloc = ((lane >> 4) << 3);
  const int cloc = lane & 15;
#pragma unroll
  for (int mi = 0; mi < 2; ++mi)
#pragma unroll
    for (int ni = 0; ni < 2; ++ni) {
      const int hcol = hTile + wn * 32 + ni * 16 + cloc;
      const float bb1 = b1[e * HDIM + hcol];
      const float bb2 = b2[e * HDIM + hcol];
#pragma unroll
      for (int r = 0; r < 8; ++r) {
        int grow = ti * BM + wm * 32 + mi * 16 + rloc + r;
        if (grow < ne) {
          float c1  = acc1[mi][ni][r] + bb1;
          float c2  = acc2[mi][ni][r] + bb2;
          float sig = __builtin_amdgcn_rcpf(1.0f + __expf(-c1));  // fast sigmoid
          h_act[(size_t)(seg + grow) * HDIM + hcol] = f32_to_bf16(c1 * sig * c2);
        }
      }
    }
}

// ---------------------------------------------------------------------------
// Stage 2: per expert:  out_part = h_act @ W3^T + b3   (rows are slots)
// ---------------------------------------------------------------------------
__global__ __launch_bounds__(256) void moe_stage2(
    const u16* __restrict__ h_act, const u16* __restrict__ w3b,
    const float* __restrict__ b3, const int* __restrict__ counts,
    const int* __restrict__ offsets, float* __restrict__ out_part) {
  const int e  = blockIdx.z;
  const int ne = counts[e];
  const int ti = blockIdx.y;
  if (ti * BM >= ne) return;
  const int cTile = blockIdx.x * BN;
  const int seg   = offsets[e];

  __shared__ u16 As[2][BM * LDSTR];
  __shared__ u16 Bs[2][BN * LDSTR];

  const int tid  = threadIdx.x;
  const int lane = tid & 31;
  const int wm   = (tid >> 5) & 3;
  const int wn   = (tid >> 5) >> 2;

  const int arow = tid >> 1;
  const int aseg = (tid & 1) << 4;
  const int gr   = ti * BM + arow;
  const int slot = seg + (gr < ne ? gr : ne - 1);
  const u16* aG  = h_act + (size_t)slot * HDIM + aseg;
  const int aoff = arow * LDSTR + aseg;

  const int brow = tid >> 2;              // 64 rows
  const int bseg = (tid & 3) << 3;        // 4 x 8 u16
  const u16* bG  = w3b + ((size_t)e * DDIM + (cTile + brow)) * HDIM + bseg;
  const int boff = brow * LDSTR + bseg;

  v8f acc[2][2] = {};

  auto load_tiles = [&](int buf, int k0) {
    const u16* a = aG + k0;
    u16* al = As[buf] + aoff;
    cp_b128(al, a); cp_b128(al + 8, a + 8);
    cp_b128(Bs[buf] + boff, bG + k0);
  };

  auto compute_chunk = [&](int buf) {
    const u16* Ab = As[buf] + (wm * 32 + (lane & 15)) * LDSTR;
    const u16* Bb = Bs[buf] + (wn * 32 + (lane & 15)) * LDSTR;
    v16bf bf[2];
#pragma unroll
    for (int ni = 0; ni < 2; ++ni) bf[ni] = load_frag(Bb + ni * 16 * LDSTR, lane);
#pragma unroll
    for (int mi = 0; mi < 2; ++mi) {
      v16bf af = load_frag(Ab + mi * 16 * LDSTR, lane);
#pragma unroll
      for (int ni = 0; ni < 2; ++ni) acc[mi][ni] = wmma_bf16(af, bf[ni], acc[mi][ni]);
    }
  };

  constexpr int NK = HDIM / BK;   // 128 chunks
  load_tiles(0, 0);
  for (int kc = 0; kc < NK; ++kc) {
    int cur = kc & 1;
    async_wait_all();
    __syncthreads();
    if (kc + 1 < NK) load_tiles(cur ^ 1, (kc + 1) * BK);
    compute_chunk(cur);
  }

  const int rloc = ((lane >> 4) << 3);
  const int cloc = lane & 15;
#pragma unroll
  for (int mi = 0; mi < 2; ++mi)
#pragma unroll
    for (int ni = 0; ni < 2; ++ni) {
      const int ccol = cTile + wn * 32 + ni * 16 + cloc;
      const float bb = b3[e * DDIM + ccol];
#pragma unroll
      for (int r = 0; r < 8; ++r) {
        int grow = ti * BM + wm * 32 + mi * 16 + rloc + r;
        if (grow < ne)
          out_part[(size_t)(seg + grow) * DDIM + ccol] = acc[mi][ni][r] + bb;
      }
    }
}

// Final combine: y[n,c] = w0*out_part[slot0,c] + w1*out_part[slot1,c]
__global__ void gather_out_kernel(const float* __restrict__ out_part,
                                  const int* __restrict__ slot_of,
                                  const float* __restrict__ tok_w,
                                  float* __restrict__ y) {
  int idx = blockIdx.x * blockDim.x + threadIdx.x;
  if (idx >= N_TOK * DDIM) return;
  int n = idx >> 10;           // /DDIM
  int c = idx & (DDIM - 1);
  int s0 = slot_of[n * 2], s1 = slot_of[n * 2 + 1];
  y[idx] = tok_w[n * 2]     * out_part[(size_t)s0 * DDIM + c] +
           tok_w[n * 2 + 1] * out_part[(size_t)s1 * DDIM + c];
}

// ---------------------------------------------------------------------------
// Launch
// ---------------------------------------------------------------------------
extern "C" void kernel_launch(void* const* d_in, const int* in_sizes, int n_in,
                              void* d_out, int out_size, void* d_ws, size_t ws_size,
                              hipStream_t stream) {
  const float* x       = (const float*)d_in[0];   // (4,1024,1024)
  const float* gate_w  = (const float*)d_in[1];   // (8,1024)
  const float* rscale  = (const float*)d_in[2];   // (1,)
  const float* w1      = (const float*)d_in[3];   // (8,4096,1024)
  const float* b1      = (const float*)d_in[4];   // (8,4096)
  const float* w2      = (const float*)d_in[5];   // (8,4096,1024)
  const float* b2      = (const float*)d_in[6];   // (8,4096)
  const float* w3      = (const float*)d_in[7];   // (8,1024,4096)
  const float* b3      = (const float*)d_in[8];   // (8,1024)
  float* y             = (float*)d_out;

  // Workspace carve-up (all segments 256-aligned by construction)
  char* p = (char*)d_ws;
  u16* xb       = (u16*)p;   p += (size_t)N_TOK * DDIM * 2;          // 8 MB
  u16* w1b      = (u16*)p;   p += (size_t)NEXP * HDIM * DDIM * 2;    // 64 MB
  u16* w2b      = (u16*)p;   p += (size_t)NEXP * HDIM * DDIM * 2;    // 64 MB
  u16* w3b      = (u16*)p;   p += (size_t)NEXP * DDIM * HDIM * 2;    // 64 MB
  u16* h_act    = (u16*)p;   p += (size_t)N_TOK * TOPK * HDIM * 2;   // 64 MB
  float* opart  = (float*)p; p += (size_t)N_TOK * TOPK * DDIM * 4;   // 32 MB
  int* counts   = (int*)p;   p += 256;
  int* offsets  = (int*)p;   p += 256;
  int* tok_e    = (int*)p;   p += (size_t)N_TOK * TOPK * 4;
  int* tok_pos  = (int*)p;   p += (size_t)N_TOK * TOPK * 4;
  float* tok_w  = (float*)p; p += (size_t)N_TOK * TOPK * 4;
  int* tok_list = (int*)p;   p += (size_t)N_TOK * TOPK * 4;
  int* slot_of  = (int*)p;   p += (size_t)N_TOK * TOPK * 4;

  zero_counts_kernel<<<1, 32, 0, stream>>>(counts);

  f2bf_kernel<<<2048, 256, 0, stream>>>(x,  xb,  N_TOK * DDIM);
  f2bf_kernel<<<8192, 256, 0, stream>>>(w1, w1b, NEXP * HDIM * DDIM);
  f2bf_kernel<<<8192, 256, 0, stream>>>(w2, w2b, NEXP * HDIM * DDIM);
  f2bf_kernel<<<8192, 256, 0, stream>>>(w3, w3b, NEXP * DDIM * HDIM);

  router_kernel<<<N_TOK / 8, 256, 0, stream>>>(x, gate_w, rscale,
                                               counts, tok_e, tok_pos, tok_w);
  offsets_kernel<<<1, 32, 0, stream>>>(counts, offsets);
  assign_slots_kernel<<<(N_TOK * TOPK + 255) / 256, 256, 0, stream>>>(
      offsets, tok_e, tok_pos, tok_list, slot_of);

  moe_stage1<<<dim3(HDIM / BN, N_TOK / BM, NEXP), 256, 0, stream>>>(
      xb, w1b, w2b, b1, b2, counts, offsets, tok_list, h_act);

  moe_stage2<<<dim3(DDIM / BN, N_TOK / BM, NEXP), 256, 0, stream>>>(
      h_act, w3b, b3, counts, offsets, opart);

  gather_out_kernel<<<(N_TOK * DDIM + 255) / 256, 256, 0, stream>>>(
      opart, slot_of, tok_w, y);
}
